// End2End_7645041787474
// MI455X (gfx1250) — matrix operand order repaired and checked by reading, not compiled, not run on hardware
//
#include <hip/hip_runtime.h>
#include <hip/hip_bf16.h>
#include <math.h>

typedef __attribute__((ext_vector_type(16))) _Float16 v16h;
typedef __attribute__((ext_vector_type(8)))  _Float16 v8h;
typedef __attribute__((ext_vector_type(8)))  float    v8f;
typedef __attribute__((ext_vector_type(4)))  int      v4i;

#define B_    8
#define L_    128
#define VF_   32128
#define V_    32100
#define D_    768
#define NPAD_ 32256      // 126 blocks * 256 cols, >= V_, zero-padded
#define EPS_  1e-8f
#define TAU_  1.0f

// ---------------------------------------------------------------------------
// CDNA5 async global->LDS copy (ASYNCcnt-tracked) + split-counter waits.
// The builtin takes AS1/AS3 pointers to a b128 (v4i) payload.
// ---------------------------------------------------------------------------
__device__ __forceinline__ void async_copy_b128(const _Float16* g, _Float16* l) {
#if __has_builtin(__builtin_amdgcn_global_load_async_to_lds_b128)
  __builtin_amdgcn_global_load_async_to_lds_b128(
      (__attribute__((address_space(1))) v4i*)g,
      (__attribute__((address_space(3))) v4i*)l, 0, 0);
#else
  unsigned lds = (unsigned)(uintptr_t)(__attribute__((address_space(3))) void*)l;
  asm volatile("global_load_async_to_lds_b128 %0, %1, off"
               :: "v"(lds), "v"((unsigned long long)(uintptr_t)g) : "memory");
#endif
}

__device__ __forceinline__ void wait_async_5() {
#if __has_builtin(__builtin_amdgcn_s_wait_asynccnt)
  __builtin_amdgcn_s_wait_asynccnt(5);
#else
  asm volatile("s_wait_asynccnt 5" ::: "memory");
#endif
}

__device__ __forceinline__ void wait_async_0() {
#if __has_builtin(__builtin_amdgcn_s_wait_asynccnt)
  __builtin_amdgcn_s_wait_asynccnt(0);
#else
  asm volatile("s_wait_asynccnt 0" ::: "memory");
#endif
}

// ---------------------------------------------------------------------------
// Kernel 1: fused (logits+gumbel)/tau -> online softmax -> argmax + scale.
// One streaming pass over 263 MB (HBM-bound, ~11us @ 23.3 TB/s).
// ---------------------------------------------------------------------------
__global__ void gumbel_argmax_k(const float* __restrict__ logits,
                                const float* __restrict__ gumbel,
                                int* __restrict__ kidx,
                                float* __restrict__ scale) {
  const int row = blockIdx.x;              // b*L + l
  const int tid = threadIdx.x;             // 256 threads = 8 waves
  const float4* lz = (const float4*)(logits + (size_t)row * VF_);
  const float4* lg = (const float4*)(gumbel + (size_t)row * VF_);
  const float invTau = 1.0f / TAU_;

  float m = -INFINITY, s = 0.f;
  int bi = 0x7fffffff;
  const int n4 = VF_ / 4;
  for (int i = tid; i < n4; i += 256) {
    float4 a = lz[i];
    float4 g = lg[i];
    float z[4] = { (a.x + g.x) * invTau, (a.y + g.y) * invTau,
                   (a.z + g.z) * invTau, (a.w + g.w) * invTau };
#pragma unroll
    for (int c = 0; c < 4; ++c) {
      float zc = z[c];
      int idx = i * 4 + c;
      if (zc > m) { s = s * expf(m - zc) + 1.f; m = zc; bi = idx; }
      else        { s += expf(zc - m); if (zc == m && idx < bi) bi = idx; }
    }
  }

  __shared__ float sm[256], ss[256];
  __shared__ int   si[256];
  sm[tid] = m; ss[tid] = s; si[tid] = bi;
  __syncthreads();
  for (int off = 128; off > 0; off >>= 1) {
    if (tid < off) {
      float m1 = sm[tid], s1 = ss[tid];  int i1 = si[tid];
      float m2 = sm[tid + off], s2 = ss[tid + off]; int i2 = si[tid + off];
      if (m2 > m1)      { sm[tid] = m2; ss[tid] = s2 + s1 * expf(m1 - m2); si[tid] = i2; }
      else if (m2 < m1) { ss[tid] = s1 + s2 * expf(m2 - m1); }
      else              { ss[tid] = s1 + s2; si[tid] = (i2 < i1) ? i2 : i1; }
    }
    __syncthreads();
  }
  if (tid == 0) {
    float y = 1.f / ss[0];               // softmax prob at the argmax
    kidx[row]  = si[0];
    scale[row] = (1.f + y) - y;          // forward value of hard + y - sg(y)
  }
}

// ---------------------------------------------------------------------------
// Kernel 2: per-batch length / psg shift / first-nonzero scan -> per-row
// (token1, coeff1) and (token2, coeff2). Also zero-inits argmax accumulators.
// ---------------------------------------------------------------------------
__global__ void psg_k(const int* __restrict__ att, const int* __restrict__ psg,
                      const int* __restrict__ kidx, const float* __restrict__ scale,
                      int* __restrict__ tok1, float* __restrict__ c1,
                      int* __restrict__ tok2, float* __restrict__ c2,
                      unsigned long long* __restrict__ out64) {
  const int b = blockIdx.x, pos = threadIdx.x;   // 128 threads
  __shared__ int s_att[L_], s_extr[L_], s_red[L_];

  int a = att[b * L_ + pos];
  s_att[pos] = a;
  s_red[pos] = a;
  __syncthreads();
  for (int off = 64; off > 0; off >>= 1) {
    if (pos < off) s_red[pos] += s_red[pos + off];
    __syncthreads();
  }
  const int len = s_red[0];
  __syncthreads();

  int keep = 1 - s_att[L_ - 1 - pos];            // 1 - fliplr(att)
  s_extr[pos] = keep * psg[b * L_ + pos];
  __syncthreads();

  int src  = (pos - len + L_) & (L_ - 1);        // (pos - shifts) mod L
  int idxv = s_extr[src];
  __syncthreads();
  s_red[pos] = (idxv != 0) ? pos : L_;
  __syncthreads();
  for (int off = 64; off > 0; off >>= 1) {
    if (pos < off) { int o = s_red[pos + off]; if (o < s_red[pos]) s_red[pos] = o; }
    __syncthreads();
  }
  const int firstnz = s_red[0];                  // cumsum(idx!=0)>0 <=> pos>=firstnz

  const int r = b * L_ + pos;
  int  k      = kidx[r];
  bool valid1 = (pos < len) && (k < V_);         // attention mask & g[..., :V]
  tok1[r] = valid1 ? k : 0;
  c1[r]   = valid1 ? scale[r] : 0.f;
  bool flag = (pos >= firstnz);
  tok2[r] = flag ? idxv : 0;
  c2[r]   = flag ? 1.f : 0.f;
  out64[r] = 0ull;
}

// ---------------------------------------------------------------------------
// Kernel 3: embeds[row] = c1*W[tok1] + c2*W[tok2]; fp32 output + normalized
// fp16 copy (rn) for the WMMA GEMM.
// ---------------------------------------------------------------------------
__global__ void embeds_k(const float* __restrict__ W,
                         const int* __restrict__ tok1, const float* __restrict__ c1,
                         const int* __restrict__ tok2, const float* __restrict__ c2,
                         float* __restrict__ out_embeds,
                         _Float16* __restrict__ rn) {
  const int row = blockIdx.x, tid = threadIdx.x; // 256 threads, 3 elems each
  const float a1 = c1[row], a2 = c2[row];
  const float* w1 = W + (size_t)tok1[row] * D_;
  const float* w2 = W + (size_t)tok2[row] * D_;
  float e[3]; float sq = 0.f;
#pragma unroll
  for (int j = 0; j < 3; ++j) {
    int d = tid + j * 256;
    float v = a1 * w1[d] + a2 * w2[d];
    e[j] = v;
    out_embeds[(size_t)row * D_ + d] = v;
    sq += v * v;
  }
  __shared__ float sred[256];
  sred[tid] = sq; __syncthreads();
  for (int off = 128; off > 0; off >>= 1) {
    if (tid < off) sred[tid] += sred[tid + off];
    __syncthreads();
  }
  const float inv = 1.f / fmaxf(sqrtf(sred[0]), EPS_);
#pragma unroll
  for (int j = 0; j < 3; ++j) {
    int d = tid + j * 256;
    rn[(size_t)row * D_ + d] = (_Float16)(e[j] * inv);
  }
}

// ---------------------------------------------------------------------------
// Kernel 4: wn = normalize(W rows) -> fp16 [NPAD_][D_], zero past V_.
// 49.5 MB fp16: resident in the 192 MB L2 for the GEMM.
// ---------------------------------------------------------------------------
__global__ void wnorm_k(const float* __restrict__ W, _Float16* __restrict__ wn) {
  const int row = blockIdx.x, tid = threadIdx.x;
  if (row >= V_) {
#pragma unroll
    for (int j = 0; j < 3; ++j) wn[(size_t)row * D_ + tid + j * 256] = (_Float16)0.f;
    return;
  }
  const float* w = W + (size_t)row * D_;
  float e[3]; float sq = 0.f;
#pragma unroll
  for (int j = 0; j < 3; ++j) { e[j] = w[tid + j * 256]; sq += e[j] * e[j]; }
  __shared__ float sred[256];
  sred[tid] = sq; __syncthreads();
  for (int off = 128; off > 0; off >>= 1) {
    if (tid < off) sred[tid] += sred[tid + off];
    __syncthreads();
  }
  const float inv = 1.f / fmaxf(sqrtf(sred[0]), EPS_);
#pragma unroll
  for (int j = 0; j < 3; ++j) {
    int d = tid + j * 256;
    wn[(size_t)row * D_ + d] = (_Float16)(e[j] * inv);
  }
}

// ---------------------------------------------------------------------------
// Kernel 5: sims = rn (1024xK) * wn^T (KxNPAD), K=768, fused argmax.
// Block: 256 thr (8 waves), tile 64M x 256N x 32K, double-buffered LDS stages
// filled with GLOBAL_LOAD_ASYNC_TO_LDS_B128 (ASYNCcnt), consumed by
// v_wmma_f32_16x16x32_f16 from ds_load fragments. Wave tile 32x64 = 8 accs.
// LDS rows padded to 80B (bank = 20r mod 64 -> 16 distinct banks/fragment).
// ---------------------------------------------------------------------------
union HFrag { v16h v; v8h h[2]; };

__device__ __forceinline__ unsigned int fkey(float f) {
  unsigned int u = __float_as_uint(f);
  return (u & 0x80000000u) ? ~u : (u | 0x80000000u);  // order-preserving
}

#define BK_   32
#define LDA_  40   // padded row length (halves) = 80B
#define KSTEPS_ (D_ / BK_)  // 24

__device__ __forceinline__ void issue_stage(const _Float16* __restrict__ rn,
                                            const _Float16* __restrict__ wn,
                                            int blockM, int blockN, int kb,
                                            _Float16 (*sAbuf)[LDA_],
                                            _Float16 (*sBbuf)[LDA_], int tid) {
  const int row = tid >> 2, chunk = tid & 3;          // 4 x b128 per 64B row
  async_copy_b128(rn + (size_t)(blockM + row) * D_ + kb + chunk * 8,
                  &sAbuf[row][chunk * 8]);
#pragma unroll
  for (int i = 0; i < 4; ++i) {
    int rb = i * 64 + row;
    async_copy_b128(wn + (size_t)(blockN + rb) * D_ + kb + chunk * 8,
                    &sBbuf[rb][chunk * 8]);
  }
}

__global__ void __launch_bounds__(256)
sims_argmax_k(const _Float16* __restrict__ rn, const _Float16* __restrict__ wn,
              unsigned long long* __restrict__ out64) {
  const int tid  = threadIdx.x;
  const int lane = tid & 31;
  const int wv   = tid >> 5;                     // wave 0..7 -> 2(M) x 4(N)
  const int half = lane >> 4;                    // 0 or 1
  const int lr   = lane & 15;
  const int mi   = (wv >> 2) * 32;               // wave M offset in block tile
  const int nj   = (wv & 3) * 64;                // wave N offset in block tile
  const int blockM = blockIdx.y * 64;
  const int blockN = blockIdx.x * 256;

  __shared__ __align__(16) _Float16 sA[2][64][LDA_];    // 10 KB
  __shared__ __align__(16) _Float16 sB[2][256][LDA_];   // 40 KB

  v8f acc[2][4] = {};

  // prologue: fill both stages (5 async ops per wave per stage)
  issue_stage(rn, wn, blockM, blockN, 0,       sA[0], sB[0], tid);
  issue_stage(rn, wn, blockM, blockN, BK_,     sA[1], sB[1], tid);

  for (int step = 0; step < KSTEPS_; ++step) {
    const int buf = step & 1;
    if (step + 2 <= KSTEPS_) wait_async_5();     // older stage done, newer in flight
    else                     wait_async_0();     // final stage: drain
    __syncthreads();                             // everyone's copies visible

    // A fragment (16x32 f16, ISA 7.12.2): lane = M=lr; V0-3: K=8h+0..7,
// V4-7: K=16+8h+0..7.  B fragment (32x16): lane = N=lr; same K order as A.
    HFrag af[2];
#pragma unroll
    for (int mt = 0; mt < 2; ++mt) {
      const _Float16* ar = &sA[buf][mi + mt * 16 + lr][0];
      af[mt].h[0] = *(const v8h*)(ar + 8 * half);
      af[mt].h[1] = *(const v8h*)(ar + 16 + 8 * half);
    }
#pragma unroll
    for (int nt = 0; nt < 4; ++nt) {
      const _Float16* br = &sB[buf][nj + nt * 16 + lr][0];
      HFrag bf;
      bf.h[0] = *(const v8h*)(br + 8 * half);
      bf.h[1] = *(const v8h*)(br + 16 + 8 * half);
      acc[0][nt] = __builtin_amdgcn_wmma_f32_16x16x32_f16(false, af[0].v, false, bf.v,
                                                          (short)0, acc[0][nt], false, false);
      acc[1][nt] = __builtin_amdgcn_wmma_f32_16x16x32_f16(false, af[1].v, false, bf.v,
                                                          (short)0, acc[1][nt], false, false);
    }

    __syncthreads();                             // done reading before overwrite
    if (step + 2 < KSTEPS_)
      issue_stage(rn, wn, blockM, blockN, (step + 2) * BK_, sA[buf], sB[buf], tid);
  }

  // Epilogue: C/D layout (VGPR v, lanes 0-15 -> row v, 16-31 -> row v+8,
  // col = lr). Per-row argmax: local over nt, butterfly over 16-lane half,
  // then one packed u64 atomicMax per row (key<<32 | ~col: ties -> min col).
#pragma unroll
  for (int mt = 0; mt < 2; ++mt) {
#pragma unroll
    for (int v = 0; v < 8; ++v) {
      float bv = -INFINITY; int bc = 0x7fffffff;
      float vals[4] = { acc[mt][0][v], acc[mt][1][v], acc[mt][2][v], acc[mt][3][v] };
#pragma unroll
      for (int nt = 0; nt < 4; ++nt) {
        int col = blockN + nj + nt * 16 + lr;
        float val = (col < V_) ? vals[nt] : -INFINITY;   // mask padded columns
        if (val > bv || (val == bv && col < bc)) { bv = val; bc = col; }
      }
#pragma unroll
      for (int off = 8; off >= 1; off >>= 1) {
        float ov = __shfl_xor(bv, off, 16);
        int   oc = __shfl_xor(bc, off, 16);
        if (ov > bv || (ov == bv && oc < bc)) { bv = ov; bc = oc; }
      }
      if (lr == 0 && bc != 0x7fffffff) {
        unsigned long long packed =
            ((unsigned long long)fkey(bv) << 32) | (unsigned int)(~(unsigned int)bc);
        atomicMax(out64 + (blockM + mi + mt * 16 + v + 8 * half), packed);
      }
    }
  }
}

// ---------------------------------------------------------------------------
// Kernel 6: unpack argmax columns; nn_idx written as float (mixed-tuple out).
// ---------------------------------------------------------------------------
__global__ void nn_finalize_k(const unsigned long long* __restrict__ out64,
                              float* __restrict__ nn_out) {
  int i = blockIdx.x * blockDim.x + threadIdx.x;
  if (i < B_ * L_) {
    unsigned long long p = out64[i];
    unsigned int col = ~(unsigned int)(p & 0xffffffffull);
    nn_out[i] = (float)(p ? col : 0u);
  }
}

// ---------------------------------------------------------------------------
extern "C" void kernel_launch(void* const* d_in, const int* in_sizes, int n_in,
                              void* d_out, int out_size, void* d_ws, size_t ws_size,
                              hipStream_t stream) {
  (void)in_sizes; (void)n_in; (void)out_size; (void)ws_size;
  const float* logits = (const float*)d_in[0];
  const float* gumbel = (const float*)d_in[1];
  const float* W      = (const float*)d_in[2];
  const int*   att    = (const int*)d_in[3];
  const int*   psg    = (const int*)d_in[4];
  float* out = (float*)d_out;
  char*  ws  = (char*)d_ws;

  // workspace layout (256-aligned), total ~51 MB
  int*   kidx  = (int*)(ws + 0);
  float* scale = (float*)(ws + 4096);
  int*   tok1  = (int*)(ws + 8192);
  float* c1    = (float*)(ws + 12288);
  int*   tok2  = (int*)(ws + 16384);
  float* c2    = (float*)(ws + 20480);
  unsigned long long* out64 = (unsigned long long*)(ws + 24576);
  _Float16* rn = (_Float16*)(ws + 32768);
  _Float16* wn = (_Float16*)(ws + 32768 + (size_t)B_ * L_ * D_ * 2);

  gumbel_argmax_k<<<B_ * L_, 256, 0, stream>>>(logits, gumbel, kidx, scale);
  psg_k<<<B_, L_, 0, stream>>>(att, psg, kidx, scale, tok1, c1, tok2, c2, out64);
  embeds_k<<<B_ * L_, 256, 0, stream>>>(W, tok1, c1, tok2, c2, out, rn);
  wnorm_k<<<NPAD_, 256, 0, stream>>>(W, wn);
  dim3 g5(NPAD_ / 256, (B_ * L_) / 64, 1);
  sims_argmax_k<<<g5, 256, 0, stream>>>(rn, wn, out64);
  nn_finalize_k<<<(B_ * L_ + 255) / 256, 256, 0, stream>>>(out64, out + (size_t)B_ * L_ * D_);
}
